// TopDownRefinement_38259568673203
// MI455X (gfx1250) — compile-verified, matrix operand order checked by reading, not run on hardware
//
#include <hip/hip_runtime.h>
#include <hip/hip_bf16.h>
#include <math.h>

#define DD 256
#define DEPTH 17
#define NNODES ((1 << DEPTH) - 1)
#define MT 32            // rows per workgroup tile

typedef __attribute__((ext_vector_type(16))) __bf16 v16bf;
typedef __attribute__((ext_vector_type(8)))  __bf16 v8bf;
typedef __attribute__((ext_vector_type(8)))  float  v8f;

#define XSTR (512 + 8)   // padded LDS row stride (halfs) for x tile
#define HSTR (256 + 8)   // padded LDS row stride (halfs) for hidden tile

__device__ __forceinline__ float gelu_exact(float x) {
    return 0.5f * x * (1.0f + erff(x * 0.7071067811865475f));
}

__device__ __forceinline__ v16bf load_afrag(const __bf16* ab) {
    v8bf alo = *(const v8bf*)ab;          // K = kb .. kb+7
    v8bf ahi = *(const v8bf*)(ab + 16);   // K = 16+kb .. 16+kb+7
    return __builtin_shufflevector(alo, ahi, 0,1,2,3,4,5,6,7,8,9,10,11,12,13,14,15);
}

// ---------------------------------------------------------------------------
// Pre-swizzle W1 (512x256) and W2 (256x256) f32 row-major into bf16 fragments
// in the native V_WMMA_16x16x32 B-operand layout:
//   blob[kt][ntile][lane][e], lane: n = ntile*16 + (lane&15), kb = (lane>>4)*8,
//   e<8 -> k = kt*32 + kb + e ; e>=8 -> k = kt*32 + 16 + kb + (e-8)
// ---------------------------------------------------------------------------
__global__ void pack_weights(const float* __restrict__ W1,
                             const float* __restrict__ W2,
                             __bf16* __restrict__ W1t,
                             __bf16* __restrict__ W2t) {
    int idx = blockIdx.x * blockDim.x + threadIdx.x;
    const int n1 = 16 * 16 * 32 * 16;   // 131072 entries (16 K-tiles)
    const int n2 = 8 * 16 * 32 * 16;    //  65536 entries ( 8 K-tiles)
    if (idx < n1) {
        int e = idx & 15, lane = (idx >> 4) & 31, nt = (idx >> 9) & 15, kt = idx >> 13;
        int n  = nt * 16 + (lane & 15);
        int kb = (lane >> 4) * 8;
        int k  = kt * 32 + ((e < 8) ? (kb + e) : (16 + kb + (e - 8)));
        W1t[idx] = (__bf16)W1[k * DD + n];
    } else if (idx < n1 + n2) {
        int j = idx - n1;
        int e = j & 15, lane = (j >> 4) & 31, nt = (j >> 9) & 15, kt = j >> 13;
        int n  = nt * 16 + (lane & 15);
        int kb = (lane >> 4) * 8;
        int k  = kt * 32 + ((e < 8) ? (kb + e) : (16 + kb + (e - 8)));
        W2t[j] = (__bf16)W2[k * DD + n];
    }
}

// ---------------------------------------------------------------------------
// One tree level: rows [start, start+m). Tile = 32 rows per workgroup.
// Fused: x=[h_v || hr[parent]] -> GELU(x@W1+b1) -> (@W2+b2) -> hr rows (f32).
// 8 waves; each wave owns a 32-wide N slice and both 16-row M blocks, so each
// B-fragment load feeds 2 WMMAs (halves L2 weight traffic vs a 16-row tile).
// ---------------------------------------------------------------------------
__global__ __launch_bounds__(256)
void level_mlp(const float* __restrict__ h,
               float* __restrict__ hr,
               const __bf16* __restrict__ W1t,
               const __bf16* __restrict__ W2t,
               const float* __restrict__ b1,
               const float* __restrict__ b2,
               const int* __restrict__ parent,
               int start, int m, int isRoot) {
    __shared__ __align__(16) __bf16 xTile[MT][XSTR];
    __shared__ __align__(16) __bf16 hTile[MT][HSTR];

    const int tid  = threadIdx.x;
    const int wave = tid >> 5;
    const int lane = tid & 31;
    const int row0 = blockIdx.x * MT;

    // ---- Stage x tile: 32 rows x 512 cols, f32 -> bf16 (parent gather in cols 256..511)
    for (int q = tid; q < MT * 128; q += 256) {
        int r  = q >> 7;
        int c4 = (q & 127) << 2;
        int lr = row0 + r;
        float4 v = make_float4(0.f, 0.f, 0.f, 0.f);
        if (lr < m) {
            int g = start + lr;
            if (c4 < 256) {
                v = *(const float4*)(h + (size_t)g * DD + c4);
            } else if (!isRoot) {
                int p = parent[g];                 // previous level, already in hr (stream order)
                v = *(const float4*)(hr + (size_t)p * DD + (c4 - 256));
            }
        }
        __bf16* dst = &xTile[r][c4];
        dst[0] = (__bf16)v.x; dst[1] = (__bf16)v.y;
        dst[2] = (__bf16)v.z; dst[3] = (__bf16)v.w;
    }
    __syncthreads();

    const int m16 = lane & 15;
    const int kg8 = (lane >> 4) * 8;
    const int nt0 = wave * 2;

    // ---- GEMM1: hidden(32x256) = x(32x512) @ W1, K = 512 in 16 steps of 32
    v8f acc00, acc01, acc10, acc11;
#pragma unroll
    for (int i = 0; i < 8; ++i) { acc00[i] = 0.f; acc01[i] = 0.f; acc10[i] = 0.f; acc11[i] = 0.f; }
#pragma unroll
    for (int kt = 0; kt < 16; ++kt) {
        v16bf a0 = load_afrag(&xTile[m16][kt * 32 + kg8]);        // rows 0..15
        v16bf a1 = load_afrag(&xTile[16 + m16][kt * 32 + kg8]);   // rows 16..31
        v16bf bA = *(const v16bf*)(W1t + ((size_t)((kt * 16 + nt0    ) * 32 + lane)) * 16);
        v16bf bB = *(const v16bf*)(W1t + ((size_t)((kt * 16 + nt0 + 1) * 32 + lane)) * 16);
        acc00 = __builtin_amdgcn_wmma_f32_16x16x32_bf16(false, a0, false, bA, (short)0, acc00, false, false);
        acc01 = __builtin_amdgcn_wmma_f32_16x16x32_bf16(false, a0, false, bB, (short)0, acc01, false, false);
        acc10 = __builtin_amdgcn_wmma_f32_16x16x32_bf16(false, a1, false, bA, (short)0, acc10, false, false);
        acc11 = __builtin_amdgcn_wmma_f32_16x16x32_bf16(false, a1, false, bB, (short)0, acc11, false, false);
    }

    // ---- Epilogue 1: bias + exact GELU -> bf16 hidden tile
    {
        int nn = wave * 32 + (lane & 15);
        int mb = (lane >> 4) * 8;
        float bb0 = b1[nn];
        float bb1 = b1[nn + 16];
#pragma unroll
        for (int r = 0; r < 8; ++r) {
            int mm = mb + r;
            hTile[mm][nn]           = (__bf16)gelu_exact(acc00[r] + bb0);
            hTile[mm][nn + 16]      = (__bf16)gelu_exact(acc01[r] + bb1);
            hTile[mm + 16][nn]      = (__bf16)gelu_exact(acc10[r] + bb0);
            hTile[mm + 16][nn + 16] = (__bf16)gelu_exact(acc11[r] + bb1);
        }
    }
    __syncthreads();

    // ---- GEMM2: out(32x256) = hidden(32x256) @ W2, K = 256 in 8 steps of 32
#pragma unroll
    for (int i = 0; i < 8; ++i) { acc00[i] = 0.f; acc01[i] = 0.f; acc10[i] = 0.f; acc11[i] = 0.f; }
#pragma unroll
    for (int kt = 0; kt < 8; ++kt) {
        v16bf a0 = load_afrag(&hTile[m16][kt * 32 + kg8]);
        v16bf a1 = load_afrag(&hTile[16 + m16][kt * 32 + kg8]);
        v16bf bA = *(const v16bf*)(W2t + ((size_t)((kt * 16 + nt0    ) * 32 + lane)) * 16);
        v16bf bB = *(const v16bf*)(W2t + ((size_t)((kt * 16 + nt0 + 1) * 32 + lane)) * 16);
        acc00 = __builtin_amdgcn_wmma_f32_16x16x32_bf16(false, a0, false, bA, (short)0, acc00, false, false);
        acc01 = __builtin_amdgcn_wmma_f32_16x16x32_bf16(false, a0, false, bB, (short)0, acc01, false, false);
        acc10 = __builtin_amdgcn_wmma_f32_16x16x32_bf16(false, a1, false, bA, (short)0, acc10, false, false);
        acc11 = __builtin_amdgcn_wmma_f32_16x16x32_bf16(false, a1, false, bB, (short)0, acc11, false, false);
    }

    // ---- Epilogue 2: bias + f32 store to hr (D layout: lane = (m>>3)*16 + n, elem = m&7)
    {
        int nn = wave * 32 + (lane & 15);
        int mb = (lane >> 4) * 8;
        float bb0 = b2[nn];
        float bb1 = b2[nn + 16];
#pragma unroll
        for (int r = 0; r < 8; ++r) {
            int mm = mb + r;
            int lr0 = row0 + mm;
            int lr1 = row0 + mm + 16;
            if (lr0 < m) {
                size_t o = (size_t)(start + lr0) * DD;
                hr[o + nn]      = acc00[r] + bb0;
                hr[o + nn + 16] = acc01[r] + bb1;
            }
            if (lr1 < m) {
                size_t o = (size_t)(start + lr1) * DD;
                hr[o + nn]      = acc10[r] + bb0;
                hr[o + nn + 16] = acc11[r] + bb1;
            }
        }
    }
}

// ---------------------------------------------------------------------------
// In-place LayerNorm: one wave32 per 256-float row, shuffle reductions.
// ---------------------------------------------------------------------------
__global__ __launch_bounds__(256)
void layernorm_rows(float* __restrict__ hr,
                    const float* __restrict__ gamma,
                    const float* __restrict__ beta,
                    int n) {
    int wave = threadIdx.x >> 5;
    int lane = threadIdx.x & 31;
    int row  = blockIdx.x * 8 + wave;
    if (row >= n) return;
    float* p = hr + (size_t)row * DD;
    int c0 = lane * 4;
    float4 v0 = *(const float4*)(p + c0);
    float4 v1 = *(const float4*)(p + 128 + c0);
    float s  = v0.x + v0.y + v0.z + v0.w + v1.x + v1.y + v1.z + v1.w;
    float ss = v0.x*v0.x + v0.y*v0.y + v0.z*v0.z + v0.w*v0.w
             + v1.x*v1.x + v1.y*v1.y + v1.z*v1.z + v1.w*v1.w;
#pragma unroll
    for (int off = 16; off > 0; off >>= 1) {
        s  += __shfl_xor(s,  off, 32);
        ss += __shfl_xor(ss, off, 32);
    }
    float mu   = s * (1.0f / 256.0f);
    float var  = ss * (1.0f / 256.0f) - mu * mu;
    float rstd = rsqrtf(var + 1e-5f);

    float4 g0 = *(const float4*)(gamma + c0);
    float4 g1 = *(const float4*)(gamma + 128 + c0);
    float4 e0 = *(const float4*)(beta + c0);
    float4 e1 = *(const float4*)(beta + 128 + c0);
    float4 o0, o1;
    o0.x = (v0.x - mu) * rstd * g0.x + e0.x;
    o0.y = (v0.y - mu) * rstd * g0.y + e0.y;
    o0.z = (v0.z - mu) * rstd * g0.z + e0.z;
    o0.w = (v0.w - mu) * rstd * g0.w + e0.w;
    o1.x = (v1.x - mu) * rstd * g1.x + e1.x;
    o1.y = (v1.y - mu) * rstd * g1.y + e1.y;
    o1.z = (v1.z - mu) * rstd * g1.z + e1.z;
    o1.w = (v1.w - mu) * rstd * g1.w + e1.w;
    *(float4*)(p + c0)       = o0;
    *(float4*)(p + 128 + c0) = o1;
}

extern "C" void kernel_launch(void* const* d_in, const int* in_sizes, int n_in,
                              void* d_out, int out_size, void* d_ws, size_t ws_size,
                              hipStream_t stream) {
    const float* h      = (const float*)d_in[0];
    const float* W1     = (const float*)d_in[1];
    const float* b1     = (const float*)d_in[2];
    const float* W2     = (const float*)d_in[3];
    const float* b2     = (const float*)d_in[4];
    const float* gamma  = (const float*)d_in[5];
    const float* beta   = (const float*)d_in[6];
    const int*   parent = (const int*)d_in[7];
    // d_in[8] = topo_order (unused; levels are implicit in heap layout)

    float*  hr  = (float*)d_out;                 // refined activations live in d_out
    __bf16* W1t = (__bf16*)d_ws;                 // 512*256 bf16 = 256 KB
    __bf16* W2t = W1t + 512 * 256;               // 256*256 bf16 = 128 KB

    // One-time weight pack into WMMA B-fragment layout (hits L2 afterwards).
    {
        int total = 16 * 16 * 32 * 16 + 8 * 16 * 32 * 16;
        pack_weights<<<(total + 255) / 256, 256, 0, stream>>>(W1, W2, W1t, W2t);
    }

    // 17 dependent levels, stream-ordered.
    for (int l = 0; l < DEPTH; ++l) {
        int start = (1 << l) - 1;
        int m     = 1 << l;
        int grid  = (m + MT - 1) / MT;
        level_mlp<<<grid, 256, 0, stream>>>(h, hr, W1t, W2t, b1, b2, parent,
                                            start, m, (l == 0) ? 1 : 0);
    }

    // Final in-place LayerNorm over all N rows.
    layernorm_rows<<<(NNODES + 7) / 8, 256, 0, stream>>>(hr, gamma, beta, NNODES);
}
